// GAFLatentExtractor_57629871177824
// MI455X (gfx1250) — matrix operand order repaired and checked by reading, not compiled, not run on hardware
//
#include <hip/hip_runtime.h>

typedef __attribute__((ext_vector_type(16))) _Float16 v16h;
typedef __attribute__((ext_vector_type(8)))  _Float16 v8h;
typedef __attribute__((ext_vector_type(8)))  float    v8f;

struct HPair { v8h lo, hi; };
union  AFrag { v16h v; HPair p; };

// ---------------------------------------------------------------------------
// fp32 -> fp16 plain convert (used for W1, k = ic*9 + tap order)
// ---------------------------------------------------------------------------
__global__ void cvt_f32_f16_kernel(const float* __restrict__ s, _Float16* __restrict__ d, int n)
{
    int i = blockIdx.x * blockDim.x + threadIdx.x;
    if (i < n) d[i] = (_Float16)s[i];
}

// ---------------------------------------------------------------------------
// W (OC, IC, 3, 3) f32 -> (OC, 9, IC) f16   (tap-major: k = tap*IC + ic)
// ---------------------------------------------------------------------------
__global__ void cvt_w_tapmajor_kernel(const float* __restrict__ s, _Float16* __restrict__ d,
                                      int OC, int IC)
{
    int j = blockIdx.x * blockDim.x + threadIdx.x;
    int n = OC * IC * 9;
    if (j >= n) return;
    int oc  = j / (9 * IC);
    int r   = j - oc * 9 * IC;
    int tap = r / IC;
    int ic  = r - tap * IC;
    d[j] = (_Float16)s[(oc * IC + ic) * 9 + tap];
}

// ---------------------------------------------------------------------------
// Wfc (K=256, N=128) f32 -> (N=128, K=256) f16  (N-major for contiguous B runs)
// ---------------------------------------------------------------------------
__global__ void cvt_wfc_t_kernel(const float* __restrict__ s, _Float16* __restrict__ d)
{
    int j = blockIdx.x * blockDim.x + threadIdx.x;   // over 128*256
    if (j >= 128 * 256) return;
    int n = j >> 8, k = j & 255;
    d[j] = (_Float16)s[k * 128 + n];
}

// ---------------------------------------------------------------------------
// GAF: x_raw (16,32,512) -> gaf fp16 (512, 2, 128, 128); block per (b,c)
// ---------------------------------------------------------------------------
__global__ __launch_bounds__(128)
void gaf_kernel(const float* __restrict__ x_raw, _Float16* __restrict__ gaf)
{
    __shared__ float xs[128], ss[128], red[128];
    const int img = blockIdx.x;
    const int t   = threadIdx.x;
    const float* src = x_raw + (size_t)img * 512;

    float v = 0.25f * (src[t*4] + src[t*4+1] + src[t*4+2] + src[t*4+3]);

    red[t] = v; __syncthreads();
    for (int s = 64; s > 0; s >>= 1) { if (t < s) red[t] = fminf(red[t], red[t+s]); __syncthreads(); }
    float mn = red[0]; __syncthreads();
    red[t] = v; __syncthreads();
    for (int s = 64; s > 0; s >>= 1) { if (t < s) red[t] = fmaxf(red[t], red[t+s]); __syncthreads(); }
    float mx = red[0]; __syncthreads();

    float xn = 2.0f * (v - mn) / (mx - mn + 1e-8f) - 1.0f;
    xn = fminf(1.0f, fmaxf(-1.0f, xn));
    float sq = sqrtf(fmaxf(0.0f, 1.0f - xn * xn));
    xs[t] = xn; ss[t] = sq;
    __syncthreads();

    uint32_t* g0 = (uint32_t*)(gaf + (size_t)img * 2 * 128 * 128 + (size_t)t * 128);
    uint32_t* g1 = g0 + (128 * 128) / 2;
    const float xi = xs[t], si = ss[t];
    #pragma unroll 4
    for (int j = 0; j < 64; ++j) {
        float xj0 = xs[2*j],   sj0 = ss[2*j];
        float xj1 = xs[2*j+1], sj1 = ss[2*j+1];
        union { uint32_t u; _Float16 h[2]; } a, b;
        a.h[0] = (_Float16)(xi * xj0 - si * sj0);
        a.h[1] = (_Float16)(xi * xj1 - si * sj1);
        b.h[0] = (_Float16)(si * xj0 - xi * sj0);
        b.h[1] = (_Float16)(si * xj1 - xi * sj1);
        g0[j] = a.u;
        g1[j] = b.u;
    }
}

// ---------------------------------------------------------------------------
// Generic implicit-GEMM conv (used only for conv1, IC=2, K=18 -> 1 WMMA/tile).
// k = ic*9 + ky*3 + kx;  LDS layout (ic, row, x).
// ---------------------------------------------------------------------------
template<int IC, int OC, int IH, int OH, int BH>
__global__ __launch_bounds__(256)
void conv_wmma_generic(const _Float16* __restrict__ in,
                       const _Float16* __restrict__ w,    // (OC, IC*9)
                       const float*    __restrict__ bias,
                       _Float16*       __restrict__ out)
{
    constexpr int IW = IH, OW = OH;
    constexpr int ROWS   = 2 * BH + 1;
    constexpr int K      = IC * 9;
    constexpr int KSTEPS = (K + 31) / 32;
    constexpr int BANDS  = OH / BH;
    constexpr int M      = BH * OW;
    constexpr int MT     = M / 16;
    constexpr int NT     = OC / 16;

    __shared__ _Float16 lds_in[IC * ROWS * IW];

    const int img  = blockIdx.x / BANDS;
    const int band = blockIdx.x % BANDS;
    const int y0   = band * (2 * BH);

    {
        const uint32_t* gsrc = (const uint32_t*)(in + (size_t)img * IC * IH * IW);
        uint32_t* ldst = (uint32_t*)lds_in;
        constexpr int W2  = IW / 2;
        constexpr int TOT = IC * ROWS * W2;
        for (int i = threadIdx.x; i < TOT; i += 256) {
            int ic  = i / (ROWS * W2);
            int rem = i - ic * ROWS * W2;
            int yr  = rem / W2;
            int xw  = rem - yr * W2;
            int y   = y0 + yr;
            ldst[i] = (y < IH) ? gsrc[(ic * IH + y) * W2 + xw] : 0u;
        }
    }
    __syncthreads();

    const int lane = threadIdx.x & 31;
    const int wave = threadIdx.x >> 5;
    const int g    = lane >> 4;
    const int nm   = lane & 15;

    for (int tile = wave; tile < MT * NT; tile += 8) {
        const int mt = tile / NT;
        const int nt = tile - mt * NT;
        const int m  = mt * 16 + nm;
        const int oy = m / OW;
        const int ox = m - oy * OW;

        v8f acc = {};
        const _Float16* wrow = w + (size_t)(nt * 16 + nm) * K;

        for (int ks = 0; ks < KSTEPS; ++ks) {
            const int kk = ks * 32;
            v16h a;
            #pragma unroll
            for (int h = 0; h < 16; ++h) {
                const int vv = h >> 1, pp = h & 1;
                const int k  = kk + ((vv < 4) ? (g * 8 + 2 * vv + pp)
                                              : (16 + g * 8 + 2 * (vv - 4) + pp));
                _Float16 val = (_Float16)0.0f;
                if (k < K) {
                    const int ic = k / 9;
                    const int r  = k - ic * 9;
                    const int ky = r / 3;
                    const int kx = r - ky * 3;
                    const int y  = 2 * oy + ky;
                    const int x  = 2 * ox + kx;
                    if (x < IW) val = lds_in[(ic * ROWS + y) * IW + x];
                }
                a[h] = val;
            }
            v16h b;
            #pragma unroll
            for (int vv = 0; vv < 8; ++vv) {
                const int k = kk + ((vv < 4) ? (g * 8 + 2 * vv) : (16 + g * 8 + 2 * (vv - 4)));
                b[2*vv]   = (k     < K) ? wrow[k]     : (_Float16)0.0f;
                b[2*vv+1] = (k + 1 < K) ? wrow[k + 1] : (_Float16)0.0f;
            }
            acc = __builtin_amdgcn_wmma_f32_16x16x32_f16(
                      false, a, false, b, (short)0, acc, false, false);
        }

        const int oc = nt * 16 + nm;
        const float bv = bias[oc];
        _Float16* obase = out + ((size_t)img * OC + oc) * (OH * OW)
                              + (size_t)band * BH * OW + mt * 16;
        #pragma unroll
        for (int r = 0; r < 8; ++r) {
            float o = acc[r] + bv;
            o = (o > 0.0f) ? o : 0.0f;
            obase[g * 8 + r] = (_Float16)o;
        }
    }
}

// ---------------------------------------------------------------------------
// Fast implicit-GEMM conv for IC % 32 == 0 (conv2/3/4).
// LDS layout (row, x, ic); K iterated tap-major (k = tap*IC + ic):
//   A fragment = two contiguous ds b128 loads, B fragment = one 32B global run.
// Each wave owns a 32x32 macro-tile (2 M-tiles x 2 N-tiles): every A fragment
// feeds 2 WMMAs and every B fragment feeds 2 WMMAs -> ~1 ds b128 + 1 global
// b128 per WMMA.  MT*NT == 32 for all instantiations => exactly 8 macro-tiles
// == 8 waves, no tile loop.
// ---------------------------------------------------------------------------
template<int IC, int OC, int IH, int OH, int BH>
__global__ __launch_bounds__(256)
void conv_wmma_cin(const _Float16* __restrict__ in,   // (512, IC, IH, IH) NCHW
                   const _Float16* __restrict__ w,    // (OC, 9, IC) tap-major fp16
                   const float*    __restrict__ bias,
                   _Float16*       __restrict__ out)  // (512, OC, OH, OH)
{
    constexpr int IW = IH, OW = OH;
    constexpr int ROWS  = 2 * BH + 1;
    constexpr int K     = IC * 9;
    constexpr int CCH   = IC / 32;          // K-chunks per tap
    constexpr int BANDS = OH / BH;
    constexpr int M     = BH * OW;
    constexpr int MT    = M / 16;
    constexpr int NT    = OC / 16;
    static_assert(MT * NT == 32, "macro-tile grid must be 8 (one per wave)");
    constexpr int NT2 = NT / 2;             // macro-tile grid is (MT/2) x NT2

    __shared__ __align__(32) _Float16 lds_in[ROWS * IW * IC];  // (row, x, ic)

    const int img  = blockIdx.x / BANDS;
    const int band = blockIdx.x % BANDS;
    const int y0   = band * (2 * BH);

    // stage band with NCHW -> (row, x, ic) transpose; zero-pad overflow row
    {
        const uint32_t* gsrc = (const uint32_t*)(in + (size_t)img * IC * IH * IW);
        constexpr int IW2 = IW / 2;
        constexpr int TOT = IC * ROWS * IW2;
        for (int i = threadIdx.x; i < TOT; i += 256) {
            int ic  = i / (ROWS * IW2);
            int rem = i - ic * ROWS * IW2;
            int yr  = rem / IW2;
            int xw  = rem - yr * IW2;
            int y   = y0 + yr;
            uint32_t dv = (y < IH) ? gsrc[(ic * IH + y) * IW2 + xw] : 0u;
            union { uint32_t u; _Float16 h[2]; } cv; cv.u = dv;
            lds_in[(yr * IW + 2 * xw    ) * IC + ic] = cv.h[0];
            lds_in[(yr * IW + 2 * xw + 1) * IC + ic] = cv.h[1];
        }
    }
    __syncthreads();

    const int lane = threadIdx.x & 31;
    const int wave = threadIdx.x >> 5;
    const int g    = lane >> 4;
    const int nm   = lane & 15;

    const int mt0 = (wave / NT2) * 2;
    const int nt0 = (wave % NT2) * 2;

    const int m0  = mt0 * 16 + nm;          // A rows, M-tile 0
    const int m1  = m0 + 16;                // A rows, M-tile 1
    const int oy0 = m0 / OW, ox0 = m0 - oy0 * OW;
    const int oy1 = m1 / OW, ox1 = m1 - oy1 * OW;

    const _Float16* wrow0 = w + (size_t)(nt0 * 16 + nm) * K;  // B column, N-tile 0
    const _Float16* wrow1 = wrow0 + (size_t)16 * K;           // B column, N-tile 1

    v8f acc00 = {}, acc01 = {}, acc10 = {}, acc11 = {};
    const v16h zero16 = {};

    for (int tap = 0; tap < 9; ++tap) {
        const int ky = tap / 3;
        const int kx = tap - 3 * ky;
        const int x0 = 2 * ox0 + kx;
        const int x1 = 2 * ox1 + kx;
        const bool valid0 = (x0 < IW);
        const bool valid1 = (x1 < IW);
        const _Float16* abase0 = lds_in + ((2 * oy0 + ky) * IW + x0) * IC;
        const _Float16* abase1 = lds_in + ((2 * oy1 + ky) * IW + x1) * IC;

        #pragma unroll
        for (int c = 0; c < CCH; ++c) {
            const int kk = tap * IC + c * 32;   // tap-major K base
            const int ko = c * 32;              // ic offset within tap

            AFrag a0, a1;
            if (valid0) {
                a0.p.lo = *(const v8h*)(abase0 + ko + g * 16);
                a0.p.hi = *(const v8h*)(abase0 + ko + g * 16 + 8);
            } else a0.v = zero16;
            if (valid1) {
                a1.p.lo = *(const v8h*)(abase1 + ko + g * 16);
                a1.p.hi = *(const v8h*)(abase1 + ko + g * 16 + 8);
            } else a1.v = zero16;

            const v16h b0 = *(const v16h*)(wrow0 + kk + g * 16);
            const v16h b1 = *(const v16h*)(wrow1 + kk + g * 16);

            acc00 = __builtin_amdgcn_wmma_f32_16x16x32_f16(
                        false, a0.v, false, b0, (short)0, acc00, false, false);
            acc01 = __builtin_amdgcn_wmma_f32_16x16x32_f16(
                        false, a0.v, false, b1, (short)0, acc01, false, false);
            acc10 = __builtin_amdgcn_wmma_f32_16x16x32_f16(
                        false, a1.v, false, b0, (short)0, acc10, false, false);
            acc11 = __builtin_amdgcn_wmma_f32_16x16x32_f16(
                        false, a1.v, false, b1, (short)0, acc11, false, false);
        }
    }

    // epilogue: bias + ReLU + fp16 stores for the 4 sub-tiles
    const int oc0 = nt0 * 16 + nm;
    const int oc1 = oc0 + 16;
    const float bv0 = bias[oc0];
    const float bv1 = bias[oc1];
    const size_t obb = (size_t)band * BH * OW;

    _Float16* ob00 = out + ((size_t)img * OC + oc0) * (OH * OW) + obb + mt0 * 16;
    _Float16* ob10 = ob00 + 16;                                   // M-tile 1, same oc0
    _Float16* ob01 = out + ((size_t)img * OC + oc1) * (OH * OW) + obb + mt0 * 16;
    _Float16* ob11 = ob01 + 16;

    #pragma unroll
    for (int r = 0; r < 8; ++r) {
        float o00 = acc00[r] + bv0; o00 = (o00 > 0.0f) ? o00 : 0.0f;
        float o01 = acc01[r] + bv1; o01 = (o01 > 0.0f) ? o01 : 0.0f;
        float o10 = acc10[r] + bv0; o10 = (o10 > 0.0f) ? o10 : 0.0f;
        float o11 = acc11[r] + bv1; o11 = (o11 > 0.0f) ? o11 : 0.0f;
        ob00[g * 8 + r] = (_Float16)o00;
        ob01[g * 8 + r] = (_Float16)o01;
        ob10[g * 8 + r] = (_Float16)o10;
        ob11[g * 8 + r] = (_Float16)o11;
    }
}

// ---------------------------------------------------------------------------
// Global average pool: h4 (512,256,8,8) fp16 -> p (512,256) fp16
// ---------------------------------------------------------------------------
__global__ __launch_bounds__(256)
void pool_kernel(const _Float16* __restrict__ h4, _Float16* __restrict__ p)
{
    const int img = blockIdx.x;
    const int c   = threadIdx.x;
    const v8h* s = (const v8h*)(h4 + ((size_t)img * 256 + c) * 64);
    float acc = 0.0f;
    #pragma unroll
    for (int i = 0; i < 8; ++i) {
        v8h v = s[i];
        #pragma unroll
        for (int h = 0; h < 8; ++h) acc += (float)v[h];
    }
    p[(size_t)img * 256 + c] = (_Float16)(acc * (1.0f / 64.0f));
}

// ---------------------------------------------------------------------------
// FC via WMMA: z(512,128) = p(512,256) @ Wfc + bfc, with wfcT (N=128, K=256)
// ---------------------------------------------------------------------------
__global__ __launch_bounds__(256)
void fc_wmma(const _Float16* __restrict__ p,
             const _Float16* __restrict__ wfcT,
             const float*    __restrict__ bfc,
             float*          __restrict__ z)
{
    const int mt   = blockIdx.x;
    const int wave = threadIdx.x >> 5;
    const int lane = threadIdx.x & 31;
    const int g    = lane >> 4;
    const int nm   = lane & 15;
    const int n    = wave * 16 + nm;

    const _Float16* arow = p    + (size_t)(mt * 16 + nm) * 256;
    const _Float16* brow = wfcT + (size_t)n * 256;

    v8f acc = {};
    #pragma unroll
    for (int ks = 0; ks < 8; ++ks) {
        const int kk = ks * 32;
        AFrag a;
        a.p.lo = *(const v8h*)(arow + kk + g * 16);
        a.p.hi = *(const v8h*)(arow + kk + g * 16 + 8);
        const v16h b = *(const v16h*)(brow + kk + g * 16);
        acc = __builtin_amdgcn_wmma_f32_16x16x32_f16(
                  false, a.v, false, b, (short)0, acc, false, false);
    }

    const float bv = bfc[n];
    #pragma unroll
    for (int r = 0; r < 8; ++r)
        z[(size_t)(mt * 16 + g * 8 + r) * 128 + n] = acc[r] + bv;
}

// ---------------------------------------------------------------------------
// out(16,128) = mean over c of z(16*32,128)
// ---------------------------------------------------------------------------
__global__ void mean_kernel(const float* __restrict__ z, float* __restrict__ out)
{
    const int i = blockIdx.x * blockDim.x + threadIdx.x;
    if (i >= 16 * 128) return;
    const int b = i >> 7, l = i & 127;
    float acc = 0.0f;
    for (int c = 0; c < 32; ++c) acc += z[(size_t)(b * 32 + c) * 128 + l];
    out[i] = acc * (1.0f / 32.0f);
}

// ---------------------------------------------------------------------------
extern "C" void kernel_launch(void* const* d_in, const int* in_sizes, int n_in,
                              void* d_out, int out_size, void* d_ws, size_t ws_size,
                              hipStream_t stream)
{
    (void)in_sizes; (void)n_in; (void)out_size; (void)ws_size;

    const float* x_raw = (const float*)d_in[0];
    const float* W1  = (const float*)d_in[1];
    const float* b1  = (const float*)d_in[2];
    const float* W2  = (const float*)d_in[3];
    const float* b2  = (const float*)d_in[4];
    const float* W3  = (const float*)d_in[5];
    const float* b3  = (const float*)d_in[6];
    const float* W4  = (const float*)d_in[7];
    const float* b4  = (const float*)d_in[8];
    const float* Wfc = (const float*)d_in[9];
    const float* bfc = (const float*)d_in[10];
    float* out = (float*)d_out;

    char* ws = (char*)d_ws;
    size_t off = 0;
    auto take = [&](size_t bytes) -> size_t {
        size_t r = off; off += (bytes + 255) & ~(size_t)255; return r;
    };

    const size_t o_w1  = take((size_t)576    * 2);   // 32*2*9   (ic-major, generic conv1)
    const size_t o_w2  = take((size_t)18432  * 2);   // 64*9*32  (tap-major)
    const size_t o_w3  = take((size_t)73728  * 2);   // 128*9*64
    const size_t o_w4  = take((size_t)294912 * 2);   // 256*9*128
    const size_t o_wfc = take((size_t)32768  * 2);   // 128*256 (N-major)
    const size_t o_gaf = take((size_t)512 * 2 * 128 * 128 * 2);   // 33.5 MB (reused as h3)
    const size_t o_h1  = take((size_t)512 * 32 * 64 * 64 * 2);    // 134 MB (reused: h4, p, z)
    const size_t o_h2  = take((size_t)512 * 64 * 32 * 32 * 2);    // 67 MB
    const size_t o_h3  = o_gaf;                         // same size (512*128*16*16*2)
    const size_t o_h4  = o_h1;                          // 16.8 MB, h1 dead after conv2
    const size_t o_p   = o_h1 + ((size_t)32 << 20);
    const size_t o_z   = o_h1 + ((size_t)64 << 20);

    // weights -> fp16 (conv2/3/4 tap-major, Wfc transposed)
    cvt_f32_f16_kernel  <<<(576    + 255) / 256, 256, 0, stream>>>(W1, (_Float16*)(ws + o_w1), 576);
    cvt_w_tapmajor_kernel<<<(18432  + 255) / 256, 256, 0, stream>>>(W2, (_Float16*)(ws + o_w2),  64,  32);
    cvt_w_tapmajor_kernel<<<(73728  + 255) / 256, 256, 0, stream>>>(W3, (_Float16*)(ws + o_w3), 128,  64);
    cvt_w_tapmajor_kernel<<<(294912 + 255) / 256, 256, 0, stream>>>(W4, (_Float16*)(ws + o_w4), 256, 128);
    cvt_wfc_t_kernel    <<<(32768  + 255) / 256, 256, 0, stream>>>(Wfc, (_Float16*)(ws + o_wfc));

    // GAF
    gaf_kernel<<<512, 128, 0, stream>>>(x_raw, (_Float16*)(ws + o_gaf));

    // encoder convs
    conv_wmma_generic<  2,  32, 128, 64, 16><<<512 * 4, 256, 0, stream>>>(
        (const _Float16*)(ws + o_gaf), (const _Float16*)(ws + o_w1), b1, (_Float16*)(ws + o_h1));
    conv_wmma_cin< 32,  64,  64, 32,  4><<<512 * 8, 256, 0, stream>>>(
        (const _Float16*)(ws + o_h1),  (const _Float16*)(ws + o_w2), b2, (_Float16*)(ws + o_h2));
    conv_wmma_cin< 64, 128,  32, 16,  4><<<512 * 4, 256, 0, stream>>>(
        (const _Float16*)(ws + o_h2),  (const _Float16*)(ws + o_w3), b3, (_Float16*)(ws + o_h3));
    conv_wmma_cin<128, 256,  16,  8,  4><<<512 * 2, 256, 0, stream>>>(
        (const _Float16*)(ws + o_h3),  (const _Float16*)(ws + o_w4), b4, (_Float16*)(ws + o_h4));

    // pool -> FC (WMMA) -> channel mean
    pool_kernel<<<512, 256, 0, stream>>>((const _Float16*)(ws + o_h4), (_Float16*)(ws + o_p));
    fc_wmma<<<32, 256, 0, stream>>>((const _Float16*)(ws + o_p),
                                    (const _Float16*)(ws + o_wfc), bfc, (float*)(ws + o_z));
    mean_kernel<<<8, 256, 0, stream>>>((const float*)(ws + o_z), out);
}
